// GraphSAGEVAE_62637803045554
// MI455X (gfx1250) — matrix operand order repaired and checked by reading, not compiled, not run on hardware
//
#include <hip/hip_runtime.h>
#include <hip/hip_bf16.h>

#define NN  8192
#define EE  262144
#define DIN 512
#define DH  256
#define DL  64

typedef __attribute__((ext_vector_type(16))) __bf16 v16bf;
typedef __attribute__((ext_vector_type(8)))  float  v8f;
typedef __attribute__((ext_vector_type(4)))  int    v4i;

// ---- CDNA5 async global->LDS path (guarded: falls back to float4 staging) ----
#if defined(__has_builtin)
#if __has_builtin(__builtin_amdgcn_global_load_async_to_lds_b128)
#define USE_ASYNC_LDS 1
#endif
#endif

#ifdef USE_ASYNC_LDS
// builtin signature (from hipcc diagnostic): param0 = v4i addrspace(1)*,
// param1 = v4i addrspace(3)*, then imm offset, imm cpol.
#define ASYNC_CP16(gsrc, ldst)                                                   \
    __builtin_amdgcn_global_load_async_to_lds_b128(                              \
        (__attribute__((address_space(1))) v4i*)(gsrc),                          \
        (__attribute__((address_space(3))) v4i*)(ldst), 0, 0)
#if __has_builtin(__builtin_amdgcn_s_wait_asynccnt)
#define WAIT_ASYNC0() __builtin_amdgcn_s_wait_asynccnt(0)
#else
#define WAIT_ASYNC0() asm volatile("s_wait_asynccnt 0" ::: "memory")
#endif
#endif

// ---------------- utility kernels ----------------

__global__ void fill_zero(float* __restrict__ p, int n) {
    int i = blockIdx.x * blockDim.x + threadIdx.x;
    if (i < n) p[i] = 0.0f;
}

__global__ void degree_kernel(const long long* __restrict__ dst, float* __restrict__ deg) {
    int e = blockIdx.x * blockDim.x + threadIdx.x;
    if (e < EE) {
        int d = ((int)dst[e]) & (NN - 1);
        atomicAdd(&deg[d], 1.0f);
    }
}

// edge-parallel scatter-sum: acc[dst] += x[src]; 4 floats per thread
__global__ void scatter_sum(const float* __restrict__ x,
                            const long long* __restrict__ src,
                            const long long* __restrict__ dst,
                            float* __restrict__ acc, int D) {
    int idx = blockIdx.x * blockDim.x + threadIdx.x;
    int chunks = D >> 2;
    if (idx >= EE * chunks) return;
    int e = idx / chunks;
    int j = (idx - e * chunks) << 2;
    int s = ((int)src[e]) & (NN - 1);
    int d = ((int)dst[e]) & (NN - 1);
    const float4 v = *(const float4*)(x + (long long)s * D + j);
    float* base = acc + (long long)d * D + j;
    atomicAdd(base + 0, v.x);
    atomicAdd(base + 1, v.y);
    atomicAdd(base + 2, v.z);
    atomicAdd(base + 3, v.w);
}

__global__ void scale_mean(float* __restrict__ acc, const float* __restrict__ deg, int D) {
    long long i = (long long)blockIdx.x * blockDim.x + threadIdx.x;
    if (i < (long long)NN * D) {
        float c = deg[i / D];
        acc[i] *= 1.0f / fmaxf(c, 1.0f);
    }
}

// ---------------- WMMA GEMM ----------------
// C[M,N] = act( A1[M,K1]@B1[K1,N] + A2[M,K2]@B2[K2,N] + bias )
// transB: B given as [N,K] row-major (z @ z^T).
// Block = 128 threads (4 wave32), tile 64x64, K step 32.
// Tiles staged to LDS as raw fp32 (async global->LDS b128 when available),
// converted fp32->bf16 at fragment build; v_wmma_f32_16x16x32_bf16, f32 acc.
__global__ void __launch_bounds__(128)
sage_gemm_wmma(const float* __restrict__ A1, const float* __restrict__ B1, int K1, int lda1, int ldb1,
               const float* __restrict__ A2, const float* __restrict__ B2, int K2, int lda2, int ldb2,
               const float* __restrict__ bias, float* __restrict__ C, int ldc,
               int transB, int relu)
{
    __shared__ float Af[64 * 32];    // A[m][k], row stride 32 (128B rows)
    __shared__ float Bsh[32 * 65];   // normal: Bk[k][n] stride 65 (pad); transB: Bt[n][k] stride 32

    const int tid  = threadIdx.x;
    const int lane = tid & 31;
    const int wave = tid >> 5;       // rows [16*wave, 16*wave+16)
    const int half = lane >> 4;      // ISA lane-group K split
    const int l15  = lane & 15;

    const long long m0 = (long long)blockIdx.y * 64;
    const long long n0 = (long long)blockIdx.x * 64;

    v8f acc[4] = {};

    for (int pass = 0; pass < 2; ++pass) {
        const float* A  = pass ? A2 : A1;
        const float* B  = pass ? B2 : B1;
        const int  K    = pass ? K2  : K1;
        const int  lda  = pass ? lda2 : lda1;
        const int  ldb  = pass ? ldb2 : ldb1;
        if (A == nullptr) continue;                    // uniform across block

        for (int k0 = 0; k0 < K; k0 += 32) {
            // ---- stage A tile 64x32 f32 (512 x 16B chunks) ----
            #pragma unroll
            for (int c = tid; c < 512; c += 128) {
                int row = c >> 3, seg = (c & 7) << 2;
#ifdef USE_ASYNC_LDS
                ASYNC_CP16(A + (m0 + row) * lda + k0 + seg, &Af[row * 32 + seg]);
#else
                *(float4*)&Af[row * 32 + seg] =
                    *(const float4*)&A[(m0 + row) * lda + k0 + seg];
#endif
            }
            // ---- stage B tile ----
            if (transB) {                              // Bt[n][k] : contiguous rows of B
                #pragma unroll
                for (int c = tid; c < 512; c += 128) {
                    int row = c >> 3, seg = (c & 7) << 2;
#ifdef USE_ASYNC_LDS
                    ASYNC_CP16(B + (n0 + row) * ldb + k0 + seg, &Bsh[row * 32 + seg]);
#else
                    *(float4*)&Bsh[row * 32 + seg] =
                        *(const float4*)&B[(n0 + row) * ldb + k0 + seg];
#endif
                }
            } else {                                   // Bk[k][n], padded stride 65
                #pragma unroll
                for (int c = tid; c < 512; c += 128) {
                    int k = c >> 4, seg = (c & 15) << 2;
                    float4 v = *(const float4*)&B[(long long)(k0 + k) * ldb + n0 + seg];
                    float* d = &Bsh[k * 65 + seg];
                    d[0] = v.x; d[1] = v.y; d[2] = v.z; d[3] = v.w;
                }
            }
#ifdef USE_ASYNC_LDS
            WAIT_ASYNC0();
#endif
            __syncthreads();

            // ---- A frag: lane m, K = half*16..+16 contiguous ----
            const float* ar = &Af[(wave * 16 + l15) * 32];
            v16bf a;
            #pragma unroll
            for (int i = 0; i < 8; ++i) a[i]     = (__bf16)ar[half * 16 + i];
            #pragma unroll
            for (int i = 0; i < 8; ++i) a[8 + i] = (__bf16)ar[half * 16 + 8 + i];

            #pragma unroll
            for (int nt = 0; nt < 4; ++nt) {
                v16bf b;
                if (transB) {                          // lane n: K = half*16..+16 contiguous
                    const float* br = &Bsh[(nt * 16 + l15) * 32 + half * 16];
                    #pragma unroll
                    for (int i = 0; i < 16; ++i) b[i] = (__bf16)br[i];
                } else {                               // gather down padded columns
                    const int n = nt * 16 + l15;
                    #pragma unroll
                    for (int i = 0; i < 16; ++i) b[i] = (__bf16)Bsh[(half * 16 + i) * 65 + n];
                }
                acc[nt] = __builtin_amdgcn_wmma_f32_16x16x32_bf16(
                    false, a, false, b, (short)0, acc[nt], false, false);
            }
            __syncthreads();
        }
    }

    // D layout: VGPR r -> row r + 8*half, col = lane&15
    #pragma unroll
    for (int nt = 0; nt < 4; ++nt) {
        long long col = n0 + nt * 16 + l15;
        float bval = bias ? bias[col] : 0.0f;
        #pragma unroll
        for (int r = 0; r < 8; ++r) {
            long long row = m0 + wave * 16 + half * 8 + r;
            float v = acc[nt][r] + bval;
            if (relu) v = fmaxf(v, 0.0f);
            C[row * ldc + col] = v;
        }
    }
}

// ---------------- reparameterization ----------------

__device__ __forceinline__ unsigned pcg_hash(unsigned v) {
    v = v * 747796405u + 2891336453u;
    unsigned w = ((v >> ((v >> 28) + 4u)) ^ v) * 277803737u;
    return (w >> 22) ^ w;
}

__global__ void reparam_kernel(const float* __restrict__ mu,
                               const float* __restrict__ logvar,
                               float* __restrict__ z) {
    int i = blockIdx.x * blockDim.x + threadIdx.x;
    if (i >= NN * DL) return;
    unsigned h1 = pcg_hash(2u * (unsigned)i + 1u);
    unsigned h2 = pcg_hash(2u * (unsigned)i + 2u);
    float u1 = ((float)h1 + 1.0f) * 2.3283064e-10f;      // (0,1]
    float u2 = (float)h2 * 2.3283064e-10f;
    float eps = sqrtf(-2.0f * __logf(u1)) * __cosf(6.2831853f * u2);
    float s = __expf(0.5f * logvar[i]);
    z[i] = mu[i] + eps * s;
}

// ---------------- launch ----------------

extern "C" void kernel_launch(void* const* d_in, const int* in_sizes, int n_in,
                              void* d_out, int out_size, void* d_ws, size_t ws_size,
                              hipStream_t stream) {
    const float*     x    = (const float*)d_in[0];
    const long long* ei   = (const long long*)d_in[1];   // int64 [2,E]
    const float*     W1l  = (const float*)d_in[2];
    const float*     b1   = (const float*)d_in[3];
    const float*     W1r  = (const float*)d_in[4];
    const float*     Wmul = (const float*)d_in[5];
    const float*     bmu  = (const float*)d_in[6];
    const float*     Wmur = (const float*)d_in[7];
    const float*     Wlvl = (const float*)d_in[8];
    const float*     blv  = (const float*)d_in[9];
    const float*     Wlvr = (const float*)d_in[10];
    const float*     Wd   = (const float*)d_in[11];
    const float*     bd   = (const float*)d_in[12];

    const long long* src = ei;
    const long long* dst = ei + EE;

    float* out     = (float*)d_out;
    float* x_recon = out;                                  // N*IN
    float* adj     = x_recon + (long long)NN * DIN;        // N*N
    float* mu      = adj + (long long)NN * NN;             // N*L
    float* logvar  = mu + (long long)NN * DL;              // N*L

    float* ws   = (float*)d_ws;
    float* deg  = ws;                                      // NN
    float* aggx = deg + NN;                                // NN*DIN
    float* h    = aggx + (long long)NN * DIN;              // NN*DH
    float* aggh = h    + (long long)NN * DH;               // NN*DH
    float* z    = aggh + (long long)NN * DH;               // NN*DL

    auto cdiv = [](long long a, long long b) { return (unsigned)((a + b - 1) / b); };

    // --- aggregate x (mean) ---
    { long long n = NN + (long long)NN * DIN;              // deg + aggx contiguous
      fill_zero<<<cdiv(n, 256), 256, 0, stream>>>(deg, (int)n); }
    degree_kernel<<<cdiv(EE, 256), 256, 0, stream>>>(dst, deg);
    scatter_sum<<<cdiv((long long)EE * (DIN / 4), 256), 256, 0, stream>>>(x, src, dst, aggx, DIN);
    scale_mean<<<cdiv((long long)NN * DIN, 256), 256, 0, stream>>>(aggx, deg, DIN);

    // --- h = relu(aggx@W1l + x@W1r + b1) ---
    { dim3 g(DH / 64, NN / 64);
      sage_gemm_wmma<<<g, 128, 0, stream>>>(aggx, W1l, DIN, DIN, DH,
                                            x,    W1r, DIN, DIN, DH,
                                            b1, h, DH, 0, 1); }

    // --- aggregate h (mean), shared by mu/logvar ---
    { long long n = (long long)NN * DH;
      fill_zero<<<cdiv(n, 256), 256, 0, stream>>>(aggh, (int)n); }
    scatter_sum<<<cdiv((long long)EE * (DH / 4), 256), 256, 0, stream>>>(h, src, dst, aggh, DH);
    scale_mean<<<cdiv((long long)NN * DH, 256), 256, 0, stream>>>(aggh, deg, DH);

    // --- mu / logvar (written straight into d_out) ---
    { dim3 g(DL / 64, NN / 64);
      sage_gemm_wmma<<<g, 128, 0, stream>>>(aggh, Wmul, DH, DH, DL,
                                            h,    Wmur, DH, DH, DL,
                                            bmu, mu, DL, 0, 0);
      sage_gemm_wmma<<<g, 128, 0, stream>>>(aggh, Wlvl, DH, DH, DL,
                                            h,    Wlvr, DH, DH, DL,
                                            blv, logvar, DL, 0, 0); }

    // --- z = mu + eps * exp(0.5*logvar) ---
    reparam_kernel<<<cdiv((long long)NN * DL, 256), 256, 0, stream>>>(mu, logvar, z);

    // --- x_recon = z @ Wd + bd ---
    { dim3 g(DIN / 64, NN / 64);
      sage_gemm_wmma<<<g, 128, 0, stream>>>(z, Wd, DL, DL, DIN,
                                            nullptr, nullptr, 0, 0, 0,
                                            bd, x_recon, DIN, 0, 0); }

    // --- adj_recon = z @ z^T  (dominant: 256 MB streamed out) ---
    { dim3 g(NN / 64, NN / 64);
      sage_gemm_wmma<<<g, 128, 0, stream>>>(z, z, DL, DL, DL,
                                            nullptr, nullptr, 0, 0, 0,
                                            nullptr, adj, NN, 1, 0); }
}
